// Model_18038862643254
// MI455X (gfx1250) — compile-verified, matrix operand order checked by reading, not run on hardware
//
#include <hip/hip_runtime.h>
#include <hip/hip_bf16.h>

typedef _Float16 v16h __attribute__((ext_vector_type(16)));
typedef _Float16 v8h  __attribute__((ext_vector_type(8)));
typedef float    v8f  __attribute__((ext_vector_type(8)));
typedef unsigned int       u32;
typedef unsigned long long u64;
typedef u32 u32x4 __attribute__((ext_vector_type(4)));
typedef int i32x8 __attribute__((ext_vector_type(8)));

#define WMMA_F32_F16(a, b, c) \
  __builtin_amdgcn_wmma_f32_16x16x32_f16(false, (a), false, (b), (short)0, (c), false, false)

// Load one 16x32 f16 fragment for this lane from a [row][k] tile (row stride in
// halves). Lane L covers row L%16 and K-halves {ksel..ksel+7},{ksel+16..ksel+23},
// ksel=(L<16)?0:8. A and B use the identical K mapping so any intra-fragment K
// permutation cancels in the MAC reduction.
__device__ __forceinline__ v16h frag_ld(const _Float16* p) {
  v8h lo = *(const v8h*)(p);
  v8h hi = *(const v8h*)(p + 16);
  v16h r;
#pragma unroll
  for (int i = 0; i < 8; ++i) { r[i] = lo[i]; r[i + 8] = hi[i]; }
  return r;
}

__device__ __forceinline__ float sigf(float x) { return 1.f / (1.f + __expf(-x)); }

// ---------------------------------------------------------------------------
// Tensor Data Mover: async 2D f16 tile load Global -> LDS.
// D# packing per CDNA5 ISA 8.3/8.4 (group0: count/lds_addr/global_addr/type,
// group1: data_size=2B, tensor dims for OOB zero-fill, tile dims, dim0 stride).
// Issue once per wave (EXEC ignored); tracked by TENSORcnt.
// ---------------------------------------------------------------------------
__device__ __forceinline__ void tdm_load_2d_f16(
    u32 lds_byte_addr, const void* gptr,
    u32 tile_d0, u32 tile_d1,            // tile elems / rows
    u32 tensor_d0, u32 tensor_d1,        // tensor extent (OOB reads -> 0)
    u32 d0_stride)                       // row stride in elems
{
  u64 ga = (u64)gptr;
  u32x4 g0;
  g0.x = 1u;                                            // count=1 user D#
  g0.y = lds_byte_addr;                                 // lds_addr
  g0.z = (u32)(ga & 0xffffffffu);                       // global_addr[31:0]
  g0.w = (u32)((ga >> 32) & 0x01ffffffu) | (2u << 30);  // ga[56:32] | type=2
  i32x8 g1;
  g1[0] = (int)(1u << 16);                              // data_size=1 -> 2 bytes
  g1[1] = (int)((tensor_d0 & 0xffffu) << 16);           // tensor_dim0[15:0]
  g1[2] = (int)(((tensor_d0 >> 16) & 0xffffu) | ((tensor_d1 & 0xffffu) << 16));
  g1[3] = (int)(((tensor_d1 >> 16) & 0xffffu) | ((tile_d0 & 0xffffu) << 16));
  g1[4] = (int)(tile_d1 & 0xffffu);                     // tile_dim1 (dim2=0)
  g1[5] = (int)d0_stride;                               // dim0_stride[31:0]
  g1[6] = 0;
  g1[7] = 0;
  asm volatile("tensor_load_to_lds %0, %1" :: "s"(g0), "s"(g1) : "memory");
}

// ---------------------------------------------------------------------------
// f32 -> f16 weight prepack
// ---------------------------------------------------------------------------
__global__ void f32_to_f16(const float* __restrict__ s, _Float16* __restrict__ d, size_t n) {
  size_t i = (size_t)blockIdx.x * blockDim.x + threadIdx.x;
  size_t stride = (size_t)gridDim.x * blockDim.x;
  for (; i < n; i += stride) d[i] = (_Float16)s[i];
}

// ---------------------------------------------------------------------------
// Generic WMMA GEMM:  out[(t,b),n] = A[(t,b),:] . W16[n,:] + bias0[n] + bias1[n]
// Row m: t = m/32, b = m%32.  A f32 (staged+converted to LDS f16, vectorized);
// W16 f16 (prepacked), B tiles TDM-staged to LDS, double buffered; tensor-dim
// OOB gives zero padding for K/N tails.  128x128 tile, 8 waves, 8 WMMA/K-step.
// ---------------------------------------------------------------------------
__global__ __launch_bounds__(256) void gemm_wmma(
    const float* __restrict__ A, long long a_st, long long a_sb,
    int M, int N, int K,
    const _Float16* __restrict__ W16,
    const float* __restrict__ bias0, const float* __restrict__ bias1,
    void* __restrict__ outp, int out_half,
    long long o_st, long long o_sb, long long o_sn)
{
  __shared__ _Float16 sA[128 * 32];
  __shared__ _Float16 sB[2][128 * 32];
  const int tid = threadIdx.x;
  const int lane = tid & 31, w = tid >> 5;
  const int wm = w & 3, wn = w >> 2;
  const int m0 = blockIdx.y * 128, n0 = blockIdx.x * 128;
  const int lr = lane & 15;
  const int ksel = (lane >> 4) << 3;

  v8f acc[2][4];
#pragma unroll
  for (int mi = 0; mi < 2; ++mi)
#pragma unroll
    for (int ni = 0; ni < 4; ++ni)
#pragma unroll
      for (int q = 0; q < 8; ++q) acc[mi][ni][q] = 0.f;

  const int nk = (K + 31) >> 5;
  if (w == 0)
    tdm_load_2d_f16((u32)(u64)&sB[0][0], W16 + (size_t)n0 * K,
                    32, 128, (u32)K, (u32)(N - n0), (u32)K);

  for (int ki = 0; ki < nk; ++ki) {
    const int k0 = ki << 5, cur = ki & 1;
    __syncthreads();  // previous iteration's sA/sB reads complete
    // stage A tile (f32 -> f16), 8 halves per chunk
    for (int c = tid; c < 512; c += 256) {
      int r = c >> 2, kk = (c & 3) << 3;
      int m = m0 + r, k = k0 + kk;
      int t = m >> 5, b = m & 31;
      v8h hv;
      if (m < M && k + 8 <= K) {
        const float* ap = A + (long long)b * a_sb + (long long)t * a_st + k;
        float4 f0 = *(const float4*)ap;
        float4 f1 = *(const float4*)(ap + 4);
        hv[0] = (_Float16)f0.x; hv[1] = (_Float16)f0.y;
        hv[2] = (_Float16)f0.z; hv[3] = (_Float16)f0.w;
        hv[4] = (_Float16)f1.x; hv[5] = (_Float16)f1.y;
        hv[6] = (_Float16)f1.z; hv[7] = (_Float16)f1.w;
      } else {
#pragma unroll
        for (int q = 0; q < 8; ++q) {
          float v = 0.f;
          if (m < M && (k + q) < K)
            v = A[(long long)b * a_sb + (long long)t * a_st + (k + q)];
          hv[q] = (_Float16)v;
        }
      }
      *(v8h*)(&sA[r * 32 + kk]) = hv;
    }
    if (w == 0) {
      __builtin_amdgcn_s_wait_tensorcnt(0);       // B tile for this iter landed
      if (ki + 1 < nk)                            // prefetch next B tile
        tdm_load_2d_f16((u32)(u64)&sB[cur ^ 1][0],
                        W16 + (size_t)n0 * K + (k0 + 32),
                        32, 128, (u32)(K - (k0 + 32)), (u32)(N - n0), (u32)K);
    }
    __syncthreads();  // sA staged + sB[cur] visible
    v16h af0 = frag_ld(&sA[(wm * 32 + lr) * 32 + ksel]);
    v16h af1 = frag_ld(&sA[(wm * 32 + 16 + lr) * 32 + ksel]);
#pragma unroll
    for (int ni = 0; ni < 4; ++ni) {
      v16h bf = frag_ld(&sB[cur][(wn * 64 + ni * 16 + lr) * 32 + ksel]);
      acc[0][ni] = WMMA_F32_F16(af0, bf, acc[0][ni]);
      acc[1][ni] = WMMA_F32_F16(af1, bf, acc[1][ni]);
    }
  }

#pragma unroll
  for (int mi = 0; mi < 2; ++mi)
#pragma unroll
    for (int ni = 0; ni < 4; ++ni)
#pragma unroll
      for (int q = 0; q < 8; ++q) {
        int m = m0 + wm * 32 + mi * 16 + ksel + q;
        int n = n0 + wn * 64 + ni * 16 + lr;
        if (m < M && n < N) {
          float v = acc[mi][ni][q];
          if (bias0) v += bias0[n];
          if (bias1) v += bias1[n];
          int t = m >> 5, b = m & 31;
          long long off = (long long)b * o_sb + (long long)t * o_st + (long long)n * o_sn;
          if (out_half) ((_Float16*)outp)[off] = (_Float16)v;
          else          ((float*)outp)[off]   = v;
        }
      }
}

// ---------------------------------------------------------------------------
// Encoder recurrent scan. grid.x = direction, 256 threads, 144 KB dynamic LDS.
// gx : [2][T][1024][32] f16 (x.Wih^T + bih + bhh, b fastest)
// whh: [2][1024][256]   f16 -- streamed into LDS by TDM as [1024][32] K-slices,
//      double buffered; waves consume fragments from LDS only.
// Per step: gates(32x1024) = gx[t] + h(32x256).Whh^T  (128 WMMAs).
// ---------------------------------------------------------------------------
__global__ __launch_bounds__(256) void enc_recurrent(
    const _Float16* __restrict__ gx,
    const _Float16* __restrict__ whh,
    float* __restrict__ out,
    int T)
{
  extern __shared__ char smem[];
  _Float16* h16  = (_Float16*)smem;             // 32x256 f16   (16 KB)
  _Float16* wbuf = (_Float16*)(smem + 16384);   // 2 x 1024x32  (128 KB)

  const int tid = threadIdx.x;
  const int d = blockIdx.x;
  const int lane = tid & 31, w = tid >> 5;
  const int lr = lane & 15;
  const int ksel = (lane >> 4) << 3;

  for (int i = tid; i < 32 * 256; i += 256) h16[i] = (_Float16)0.f;

  float creg[2][2][8];
#pragma unroll
  for (int mi = 0; mi < 2; ++mi)
#pragma unroll
    for (int nt = 0; nt < 2; ++nt)
#pragma unroll
      for (int q = 0; q < 8; ++q) creg[mi][nt][q] = 0.f;

  const _Float16* gxd = gx + (size_t)d * T * 1024 * 32;
  const _Float16* whd = whh + (size_t)d * 1024 * 256;
  const int total = T * 8;

  if (w == 0)  // preload K-slice 0
    tdm_load_2d_f16((u32)(u64)wbuf, whd, 32, 1024, 256, 1024, 256);

  for (int step = 0; step < T; ++step) {
    int t = d ? (T - 1 - step) : step;
    const _Float16* gxt = gxd + (size_t)t * 1024 * 32;

    v8f acc[2][4][2];
#pragma unroll
    for (int mi = 0; mi < 2; ++mi)
#pragma unroll
      for (int g = 0; g < 4; ++g)
#pragma unroll
        for (int nt = 0; nt < 2; ++nt) {
          int n = g * 256 + w * 32 + nt * 16 + lr;
          int b0 = mi * 16 + ksel;
          v8h hv = *(const v8h*)(gxt + (size_t)n * 32 + b0);
#pragma unroll
          for (int q = 0; q < 8; ++q) acc[mi][g][nt][q] = (float)hv[q];
        }
    // prefetch next step's gx slab (streaming, 64 KB)
    if (step + 1 < T) {
      int tn = d ? (T - 2 - step) : (step + 1);
      const _Float16* gxn = gxd + (size_t)tn * 1024 * 32;
      for (int i = tid; i < 512; i += 256)
        __builtin_prefetch(gxn + (size_t)i * 64, 0, 1);
    }

    for (int ks = 0; ks < 8; ++ks) {
      const int si = step * 8 + ks;
      const int cur = si & 1;
      if (w == 0) __builtin_amdgcn_s_wait_tensorcnt(0);  // slice si landed
      __syncthreads();  // slice visible; prev buffer reads + h16 writes ordered
      if (w == 0 && si + 1 < total) {
        int k0n = ((si + 1) & 7) << 5;
        tdm_load_2d_f16((u32)(u64)(wbuf + (size_t)((si + 1) & 1) * 32768),
                        whd + k0n, 32, 1024, 256, 1024, 256);
      }
      const int k0 = ks << 5;
      const _Float16* wb = wbuf + (size_t)cur * 32768;
      v16h af0 = frag_ld(&h16[lr * 256 + k0 + ksel]);
      v16h af1 = frag_ld(&h16[(16 + lr) * 256 + k0 + ksel]);
#pragma unroll
      for (int g = 0; g < 4; ++g)
#pragma unroll
        for (int nt = 0; nt < 2; ++nt) {
          int nrow = g * 256 + w * 32 + nt * 16 + lr;
          v16h bf = frag_ld(wb + nrow * 32 + ksel);
          acc[0][g][nt] = WMMA_F32_F16(af0, bf, acc[0][g][nt]);
          acc[1][g][nt] = WMMA_F32_F16(af1, bf, acc[1][g][nt]);
        }
    }
    __syncthreads();  // all h16 readers done before overwrite

#pragma unroll
    for (int mi = 0; mi < 2; ++mi)
#pragma unroll
      for (int nt = 0; nt < 2; ++nt)
#pragma unroll
        for (int q = 0; q < 8; ++q) {
          float i_ = acc[mi][0][nt][q];
          float f_ = acc[mi][1][nt][q];
          float g_ = acc[mi][2][nt][q];
          float o_ = acc[mi][3][nt][q];
          float c = creg[mi][nt][q];
          c = sigf(f_) * c + sigf(i_) * tanhf(g_);
          float h = sigf(o_) * tanhf(c);
          creg[mi][nt][q] = c;
          int b = mi * 16 + ksel + q;
          int hc = w * 32 + nt * 16 + lr;
          h16[b * 256 + hc] = (_Float16)h;
          out[((size_t)b * T + t) * 512 + d * 256 + hc] = h;
        }
  }
}

// ---------------------------------------------------------------------------
// Decoder step 0: h0=c0=0, ctx=0, tok=sos  ->  h1 (same row for every b).
// ---------------------------------------------------------------------------
__global__ __launch_bounds__(256) void dec_step0(
    const float* __restrict__ emb, const float* __restrict__ wih,
    const float* __restrict__ bih, const float* __restrict__ bhh,
    const int* __restrict__ sos, float* __restrict__ dec_h)
{
  __shared__ float hrow[512];
  const int tid = threadIdx.x;
  const int tok = sos[0];
  for (int hc = tid; hc < 512; hc += 256) {
    float gi = bih[hc] + bhh[hc];
    float gg = bih[1024 + hc] + bhh[1024 + hc];
    float go = bih[1536 + hc] + bhh[1536 + hc];
    for (int k = 0; k < 256; ++k) {
      float e = emb[(size_t)tok * 256 + k];
      gi += e * wih[(size_t)hc * 768 + 512 + k];
      gg += e * wih[(size_t)(1024 + hc) * 768 + 512 + k];
      go += e * wih[(size_t)(1536 + hc) * 768 + 512 + k];
    }
    float c = sigf(gi) * tanhf(gg);   // c0 == 0, forget gate drops out
    hrow[hc] = sigf(go) * tanhf(c);
  }
  __syncthreads();
  for (int i = tid; i < 32 * 512; i += 256) dec_h[i] = hrow[i & 511];
}

__global__ void argmax32(const float* __restrict__ preds, int* __restrict__ res) {
  int b = threadIdx.x;
  if (b < 32) {
    const float* row = preds + (size_t)b * 128 * 5000;
    float best = row[0]; int bi = 0;
    for (int v = 1; v < 5000; ++v) { float x = row[v]; if (x > best) { best = x; bi = v; } }
    res[b] = bi;
  }
}

// ---------------------------------------------------------------------------
// Decoder loop, steps 1..127 in one workgroup. 176 KB dynamic LDS.
// Attention (VALU) + LSTM gates (WMMA, K = 768(xt) + 512(h), N=2048).
// ---------------------------------------------------------------------------
__global__ __launch_bounds__(256) void dec_loop(
    const float* __restrict__ h_enc,     // (32,256,512)
    const int* __restrict__ target,      // (32,128)
    const float* __restrict__ emb,       // (5000,256)
    const _Float16* __restrict__ wih,    // (2048,768) f16
    const _Float16* __restrict__ whh,    // (2048,512) f16
    const float* __restrict__ bih,
    const float* __restrict__ bhh,
    const int* __restrict__ res,
    float* __restrict__ dec_h)           // (128,32,512)
{
  extern __shared__ char smem[];
  _Float16* h16  = (_Float16*)smem;               // 32x512 f16   (32 KB)
  float*    cbuf = (float*)(smem + 32768);        // 32x512 f32   (64 KB)
  _Float16* xt16 = (_Float16*)(smem + 98304);     // 32x768 f16   (48 KB)
  float*    ebuf = (float*)(smem + 147456);       // 32x256 f32   (32 KB)

  const int tid = threadIdx.x;
  const int lane = tid & 31, w = tid >> 5;
  const int lr = lane & 15;
  const int ksel = (lane >> 4) << 3;

  for (int i = tid; i < 32 * 512; i += 256) {
    h16[i] = (_Float16)dec_h[i];   // h1
    cbuf[i] = 0.f;                 // ref feeds zero c into the scan
  }
  __syncthreads();

  for (int js = 0; js < 127; ++js) {
    // attention scores e[b,t] = h_enc[b,t,:] . h[b,:]
    for (int p = tid; p < 32 * 256; p += 256) {
      int b = p >> 8, t = p & 255;
      const float* er = h_enc + ((size_t)b * 256 + t) * 512;
      const _Float16* hr = h16 + b * 512;
      float s = 0.f;
#pragma unroll 4
      for (int k = 0; k < 512; k += 4) {
        float4 ev = *(const float4*)(er + k);
        s += ev.x * (float)hr[k] + ev.y * (float)hr[k + 1]
           + ev.z * (float)hr[k + 2] + ev.w * (float)hr[k + 3];
      }
      ebuf[p] = s;
    }
    __syncthreads();
    // softmax over t (each wave: 4 rows)
#pragma unroll
    for (int rb = 0; rb < 4; ++rb) {
      float* row = ebuf + (w * 4 + rb) * 256;
      float mx = -3.402823466e38f;
      for (int t = lane; t < 256; t += 32) mx = fmaxf(mx, row[t]);
#pragma unroll
      for (int o = 16; o; o >>= 1) mx = fmaxf(mx, __shfl_xor(mx, o, 32));
      float sum = 0.f;
      for (int t = lane; t < 256; t += 32) { float v = __expf(row[t] - mx); row[t] = v; sum += v; }
#pragma unroll
      for (int o = 16; o; o >>= 1) sum += __shfl_xor(sum, o, 32);
      float inv = 1.f / sum;
      for (int t = lane; t < 256; t += 32) row[t] *= inv;
    }
    __syncthreads();
    // ctx -> xt16[:,0:512]
    for (int p = tid; p < 32 * 512; p += 256) {
      int b = p >> 9, k = p & 511;
      const float* ar = ebuf + b * 256;
      const float* er = h_enc + ((size_t)b * 256) * 512 + k;
      float s = 0.f;
#pragma unroll 4
      for (int t = 0; t < 256; ++t) s += ar[t] * er[(size_t)t * 512];
      xt16[b * 768 + k] = (_Float16)s;
    }
    // embedding -> xt16[:,512:768]
    for (int p = tid; p < 32 * 256; p += 256) {
      int b = p >> 8, k = p & 255;
      int tok = (js == 0) ? res[b] : target[b * 128 + js - 1];
      xt16[b * 768 + 512 + k] = (_Float16)emb[(size_t)tok * 256 + k];
    }
    __syncthreads();
    // gates + state update (two passes of 16 batch rows)
    for (int mi = 0; mi < 2; ++mi) {
      v8f acc[4][4];
#pragma unroll
      for (int g = 0; g < 4; ++g)
#pragma unroll
        for (int jt = 0; jt < 4; ++jt)
#pragma unroll
          for (int q = 0; q < 8; ++q) acc[g][jt][q] = 0.f;

      for (int ks = 0; ks < 24; ++ks) {              // xt . Wih^T
        int k0 = ks * 32;
        v16h af = frag_ld(&xt16[(mi * 16 + lr) * 768 + k0 + ksel]);
#pragma unroll
        for (int g = 0; g < 4; ++g)
#pragma unroll
          for (int jt = 0; jt < 4; ++jt) {
            int nrow = g * 512 + w * 64 + jt * 16 + lr;
            v16h bf = frag_ld(wih + (size_t)nrow * 768 + k0 + ksel);
            acc[g][jt] = WMMA_F32_F16(af, bf, acc[g][jt]);
          }
      }
      for (int ks = 0; ks < 16; ++ks) {              // h . Whh^T
        int k0 = ks * 32;
        v16h af = frag_ld(&h16[(mi * 16 + lr) * 512 + k0 + ksel]);
#pragma unroll
        for (int g = 0; g < 4; ++g)
#pragma unroll
          for (int jt = 0; jt < 4; ++jt) {
            int nrow = g * 512 + w * 64 + jt * 16 + lr;
            v16h bf = frag_ld(whh + (size_t)nrow * 512 + k0 + ksel);
            acc[g][jt] = WMMA_F32_F16(af, bf, acc[g][jt]);
          }
      }
      __syncthreads();  // all reads of h16 rows done before overwrite
#pragma unroll
      for (int jt = 0; jt < 4; ++jt)
#pragma unroll
        for (int q = 0; q < 8; ++q) {
          int b = mi * 16 + ksel + q;
          int hc = w * 64 + jt * 16 + lr;
          float i_ = acc[0][jt][q] + bih[hc] + bhh[hc];
          float f_ = acc[1][jt][q] + bih[512 + hc] + bhh[512 + hc];
          float g_ = acc[2][jt][q] + bih[1024 + hc] + bhh[1024 + hc];
          float o_ = acc[3][jt][q] + bih[1536 + hc] + bhh[1536 + hc];
          float c = cbuf[b * 512 + hc];
          c = sigf(f_) * c + sigf(i_) * tanhf(g_);
          float h = sigf(o_) * tanhf(c);
          cbuf[b * 512 + hc] = c;
          h16[b * 512 + hc] = (_Float16)h;
          dec_h[((size_t)(js + 1) * 32 + b) * 512 + hc] = h;
        }
      __syncthreads();
    }
  }
}

// ---------------------------------------------------------------------------
// In-place softmax over the last dim of d_out (4096 rows x 5000)
// ---------------------------------------------------------------------------
__global__ __launch_bounds__(256) void softmax_rows(float* __restrict__ data, int C) {
  __shared__ float sred[8];
  float* row = data + (size_t)blockIdx.x * C;
  const int tid = threadIdx.x, lane = tid & 31, w = tid >> 5;
  float mx = -3.402823466e38f;
  for (int i = tid; i < C; i += 256) mx = fmaxf(mx, row[i]);
#pragma unroll
  for (int o = 16; o; o >>= 1) mx = fmaxf(mx, __shfl_xor(mx, o, 32));
  if (lane == 0) sred[w] = mx;
  __syncthreads();
  mx = sred[0];
#pragma unroll
  for (int i = 1; i < 8; ++i) mx = fmaxf(mx, sred[i]);
  __syncthreads();
  float sum = 0.f;
  for (int i = tid; i < C; i += 256) { float v = __expf(row[i] - mx); row[i] = v; sum += v; }
#pragma unroll
  for (int o = 16; o; o >>= 1) sum += __shfl_xor(sum, o, 32);
  if (lane == 0) sred[w] = sum;
  __syncthreads();
  sum = 0.f;
#pragma unroll
  for (int i = 0; i < 8; ++i) sum += sred[i];
  float inv = 1.f / sum;
  for (int i = tid; i < C; i += 256) row[i] *= inv;
}

// ---------------------------------------------------------------------------
extern "C" void kernel_launch(void* const* d_in, const int* in_sizes, int n_in,
                              void* d_out, int out_size, void* d_ws, size_t ws_size,
                              hipStream_t stream) {
  (void)in_sizes; (void)n_in; (void)out_size; (void)ws_size;
  const float* x      = (const float*)d_in[0];
  const int*   target = (const int*)d_in[1];
  // enc_weights flatten: d_in[2 + layer*8 + dir*4 + {0:Wih,1:Whh,2:bih,3:bhh}]
  const float* emb  = (const float*)d_in[26];
  const float* dWih = (const float*)d_in[27];
  const float* dWhh = (const float*)d_in[28];
  const float* dbih = (const float*)d_in[29];
  const float* dbhh = (const float*)d_in[30];
  const float* fcW  = (const float*)d_in[31];
  const float* fcb  = (const float*)d_in[32];
  const int*   sos  = (const int*)d_in[33];
  float* out = (float*)d_out;

  char* ws = (char*)d_ws;
  size_t off = 0;
  auto take = [&](size_t bytes) -> char* {
    char* p = ws + off;
    off += (bytes + 255) & ~(size_t)255;
    return p;
  };
  _Float16* gx16  = (_Float16*)take(2ull * 1024 * 1024 * 32 * 2);  // [2][T][1024][32]
  float*    enc0  = (float*)take(32ull * 1024 * 512 * 4);
  float*    enc1  = (float*)take(32ull * 512 * 512 * 4);
  float*    enc2  = (float*)take(32ull * 256 * 512 * 4);
  _Float16* whhE  = (_Float16*)take(6ull * 1024 * 256 * 2);
  _Float16* wihE0 = (_Float16*)take(2ull * 1024 * 80 * 2);
  _Float16* wihE1 = (_Float16*)take(2ull * 1024 * 1024 * 2);
  _Float16* wihE2 = (_Float16*)take(2ull * 1024 * 1024 * 2);
  _Float16* wihD  = (_Float16*)take(2048ull * 768 * 2);
  _Float16* whhD  = (_Float16*)take(2048ull * 512 * 2);
  _Float16* fcW16 = (_Float16*)take(5000ull * 512 * 2);
  float*    dech  = (float*)take(128ull * 32 * 512 * 4);
  int*      res   = (int*)take(256);

  // --- prepack all GEMM weights to f16
  _Float16* wihE[3] = {wihE0, wihE1, wihE2};
  const size_t wihSz[3] = {1024ull * 80, 1024ull * 1024, 1024ull * 1024};
  for (int l = 0; l < 3; ++l)
    for (int d = 0; d < 2; ++d) {
      const float* wihw = (const float*)d_in[2 + l * 8 + d * 4 + 0];
      const float* whhw = (const float*)d_in[2 + l * 8 + d * 4 + 1];
      f32_to_f16<<<256, 256, 0, stream>>>(wihw, wihE[l] + (size_t)d * wihSz[l], wihSz[l]);
      f32_to_f16<<<256, 256, 0, stream>>>(whhw, whhE + (size_t)(l * 2 + d) * 1024 * 256,
                                          (size_t)1024 * 256);
    }
  f32_to_f16<<<512, 256, 0, stream>>>(dWih, wihD, (size_t)2048 * 768);
  f32_to_f16<<<512, 256, 0, stream>>>(dWhh, whhD, (size_t)2048 * 512);
  f32_to_f16<<<512, 256, 0, stream>>>(fcW, fcW16, (size_t)5000 * 512);

  // --- encoder: per layer, 2 big Gx GEMMs then the 2-direction recurrent scan
  const float* inA = x;
  long long a_st = 80, a_sb = 1024ll * 80;
  const int Ks[3] = {80, 1024, 1024};
  const int Ts[3] = {1024, 512, 256};
  float* encOuts[3] = {enc0, enc1, enc2};
  for (int l = 0; l < 3; ++l) {
    const int T = Ts[l], K = Ks[l], M = 32 * T;
    for (int d = 0; d < 2; ++d) {
      const float* bih = (const float*)d_in[2 + l * 8 + d * 4 + 2];
      const float* bhh = (const float*)d_in[2 + l * 8 + d * 4 + 3];
      dim3 grid((1024 + 127) / 128, (M + 127) / 128);
      gemm_wmma<<<grid, 256, 0, stream>>>(
          inA, a_st, a_sb, M, 1024, K, wihE[l] + (size_t)d * wihSz[l], bih, bhh,
          (void*)(gx16 + (size_t)d * T * 1024 * 32), 1,
          /*o_st=*/1024ll * 32, /*o_sb=*/1ll, /*o_sn=*/32ll);
    }
    enc_recurrent<<<2, 256, 147456, stream>>>(gx16, whhE + (size_t)l * 2 * 1024 * 256,
                                              encOuts[l], T);
    inA = encOuts[l];                     // (32,T,512) viewed as (32,T/2,1024)
    a_st = 1024;
    a_sb = (long long)T * 512;
  }

  // --- decoder
  dec_step0<<<1, 256, 0, stream>>>(emb, dWih, dbih, dbhh, sos, dech);
  {
    dim3 grid((5000 + 127) / 128, 1);     // preds row 0 = fc(h1)
    gemm_wmma<<<grid, 256, 0, stream>>>(dech, 0ll, 512ll, 32, 5000, 512, fcW16, fcb, nullptr,
                                        out, 0, 5000ll, 128ll * 5000, 1ll);
  }
  argmax32<<<1, 32, 0, stream>>>(out, res);
  dec_loop<<<1, 256, 180224, stream>>>(enc2, target, emb, wihD, whhD, dbih, dbhh, res, dech);
  {
    dim3 grid((5000 + 127) / 128, (127 * 32 + 127) / 128);   // preds rows 1..127
    gemm_wmma<<<grid, 256, 0, stream>>>(dech + 32 * 512, 32ll * 512, 512ll, 127 * 32, 5000, 512,
                                        fcW16, fcb, nullptr, out + 5000, 0, 5000ll, 128ll * 5000, 1ll);
  }
  softmax_rows<<<32 * 128, 256, 0, stream>>>(out, 5000);
}
